// GNNFeatureExtractor_66881230734147
// MI455X (gfx1250) — matrix operand order repaired and checked by reading, not compiled, not run on hardware
//
#include <hip/hip_runtime.h>
#include <hip/hip_bf16.h>

typedef _Float16 v16h __attribute__((ext_vector_type(16)));
typedef _Float16 v8h  __attribute__((ext_vector_type(8)));
typedef float    v8f  __attribute__((ext_vector_type(8)));

#define BSZ   512
#define NNODE 400
#define MROWS (BSZ * NNODE)   // 204800
#define EDIM  128

__device__ __forceinline__ v16h cat16(v8h lo, v8h hi) {
    return __builtin_shufflevector(lo, hi, 0,1,2,3,4,5,6,7,8,9,10,11,12,13,14,15);
}

// ---------------------------------------------------------------------------
// Kernel A: build feat16 (B,400,32 f16; cols 0..6 main counts, 7..13 agv
// counts, 14 dist, 15..31 zero pad) and seed[b].
// ---------------------------------------------------------------------------
__global__ __launch_bounds__(256)
void featKernel(const float* __restrict__ agvs, const float* __restrict__ nodes,
                _Float16* __restrict__ feat16, int* __restrict__ seed) {
    __shared__ float snodes[2 * NNODE];
    __shared__ float sfeat[NNODE * 15];
    const int b = blockIdx.x;
    const int tid = threadIdx.x;

    for (int i = tid; i < 2 * NNODE; i += 256) snodes[i] = nodes[i];
    for (int i = tid; i < NNODE * 15; i += 256) sfeat[i] = 0.0f;
    __syncthreads();

    if (tid < 70) {                       // nearest-node counts (10 agvs x 7 coords)
        const int a = tid / 7, k = tid % 7;
        const float x = agvs[b * 160 + a * 16 + 2 + 2 * k];
        const float y = agvs[b * 160 + a * 16 + 3 + 2 * k];
        int best = 0; float bd = 3.0e38f;
        for (int n = 0; n < NNODE; ++n) {
            const float dx = x - snodes[2 * n], dy = y - snodes[2 * n + 1];
            const float d = dx * dx + dy * dy;
            if (d < bd) { bd = d; best = n; }
        }
        const int col = (a == 0) ? k : 7 + k;
        atomicAdd(&sfeat[best * 15 + col], 1.0f);
    }
    if (tid == 70) {                      // seed = nearest(agvs[b,0,4:6])
        const float x = agvs[b * 160 + 4], y = agvs[b * 160 + 5];
        int best = 0; float bd = 3.0e38f;
        for (int n = 0; n < NNODE; ++n) {
            const float dx = x - snodes[2 * n], dy = y - snodes[2 * n + 1];
            const float d = dx * dx + dy * dy;
            if (d < bd) { bd = d; best = n; }
        }
        seed[b] = best;
    }
    __syncthreads();

    const float mx = agvs[b * 160 + 6], my = agvs[b * 160 + 7];
    for (int n = tid; n < NNODE; n += 256) {
        const float dx = snodes[2 * n] - mx, dy = snodes[2 * n + 1] - my;
        sfeat[n * 15 + 14] = sqrtf(dx * dx + dy * dy);
    }
    __syncthreads();

    for (int i = tid; i < NNODE * 32; i += 256) {
        const int n = i >> 5, k = i & 31;
        const float v = (k < 15) ? sfeat[n * 15 + k] : 0.0f;
        feat16[((size_t)b * NNODE + n) * 32 + k] = (_Float16)v;
    }
}

// ---------------------------------------------------------------------------
// Kernel B: GEMM1  hrelu = relu(feat @ W0 + b0)   (204800x32pad)x(32x256)
// block = 64 rows x 256 cols, 8 waves (4 row strips x 2 col halves)
// ---------------------------------------------------------------------------
__global__ __launch_bounds__(256)
void gemm1Kernel(const _Float16* __restrict__ A, const float* __restrict__ W0,
                 const float* __restrict__ b0, _Float16* __restrict__ out) {
    __shared__ alignas(16) _Float16 sW[256 * 32];   // [n][k], K padded to 32
    const int tid = threadIdx.x;
    for (int i = tid; i < 256 * 32; i += 256) {
        const int n = i >> 5, k = i & 31;
        sW[i] = (_Float16)((k < 15) ? W0[k * 256 + n] : 0.0f);
    }
    __syncthreads();

    const int w = tid >> 5, lane = tid & 31;
    const int m = lane & 15, hi = lane >> 4;
    const int rowBase = blockIdx.x * 64 + (w & 3) * 16;
    const int colBase = (w >> 2) * 128;

    const size_t abase = (size_t)(rowBase + m) * 32;
    v8h alo = *(const v8h*)(A + abase + hi * 8);
    v8h ahi = *(const v8h*)(A + abase + 16 + hi * 8);
    v16h afrag = cat16(alo, ahi);

#pragma unroll
    for (int ct = 0; ct < 8; ++ct) {
        const int n0 = colBase + ct * 16;
        const int n = n0 + m;
        const _Float16* wp = sW + n * 32 + hi * 8;
        v16h bfrag = cat16(*(const v8h*)wp, *(const v8h*)(wp + 16));
        v8f acc = {};
        acc = __builtin_amdgcn_wmma_f32_16x16x32_f16(false, afrag, false, bfrag,
                                                     (short)0, acc, false, false);
        const float bias = b0[n];
#pragma unroll
        for (int v = 0; v < 8; ++v) {
            const int row = rowBase + v + hi * 8;
            float val = acc[v] + bias;
            val = fmaxf(val, 0.0f);
            out[(size_t)row * 256 + n] = (_Float16)val;
        }
    }
}

// ---------------------------------------------------------------------------
// BN stats per node: mean/var over (B, C); alpha = g*rsqrt(var+eps),
// beta = be - mean*alpha
// ---------------------------------------------------------------------------
__global__ __launch_bounds__(256)
void bnStatsKernel(const _Float16* __restrict__ X, int C,
                   const float* __restrict__ g, const float* __restrict__ be,
                   float* __restrict__ alpha, float* __restrict__ beta) {
    __shared__ float rs[256], rq[256];
    const int n = blockIdx.x, tid = threadIdx.x;
    const int total = BSZ * C;
    float s = 0.0f, s2 = 0.0f;
    for (int i = tid; i < total; i += 256) {
        const int b = i / C, c = i % C;
        const float v = (float)X[((size_t)b * NNODE + n) * C + c];
        s += v; s2 += v * v;
    }
    rs[tid] = s; rq[tid] = s2;
    __syncthreads();
    for (int st = 128; st > 0; st >>= 1) {
        if (tid < st) { rs[tid] += rs[tid + st]; rq[tid] += rq[tid + st]; }
        __syncthreads();
    }
    if (tid == 0) {
        const float mean = rs[0] / (float)total;
        const float var = rq[0] / (float)total - mean * mean;
        const float a = g[n] * rsqrtf(var + 1e-5f);
        alpha[n] = a;
        beta[n] = be[n] - mean * a;
    }
}

// ---------------------------------------------------------------------------
// Column sums of W1 (256x128) and Wc0 (128x128) for the BN fold.
// ---------------------------------------------------------------------------
__global__ __launch_bounds__(128)
void colsumKernel(const float* __restrict__ W1, const float* __restrict__ Wc0,
                  float* __restrict__ csumW1, float* __restrict__ csumWc0) {
    const int j = threadIdx.x;
    float s = 0.0f;
    for (int k = 0; k < 256; ++k) s += W1[k * 128 + j];
    csumW1[j] = s;
    s = 0.0f;
    for (int k = 0; k < 128; ++k) s += Wc0[k * 128 + j];
    csumWc0[j] = s;
}

// ---------------------------------------------------------------------------
// Generic GEMM: out(M x 128) = A(M x KDIM f16) @ W(KDIM x 128 f32->f16 LDS)
// EPI 0: relu(alpha[node]*acc + beta[node]*csum[col] + b1[col])   (GEMM2)
// EPI 1: alpha[node]*acc + beta[node]*csum[col]                   (GC layer 0)
// EPI 2: acc                                                      (GC layer 1)
// block = 128 rows, 8 waves of 16 rows x 128 cols, K staged in 32-chunks.
// ---------------------------------------------------------------------------
template <int KDIM, int EPI>
__global__ __launch_bounds__(256)
void gemmKKernel(const _Float16* __restrict__ A, const float* __restrict__ W,
                 const float* __restrict__ b1, const float* __restrict__ alpha,
                 const float* __restrict__ beta, const float* __restrict__ csum,
                 _Float16* __restrict__ out) {
    __shared__ alignas(16) _Float16 sW[128 * 32];   // [n][k] chunk
    const int tid = threadIdx.x;
    const int w = tid >> 5, lane = tid & 31;
    const int m = lane & 15, hi = lane >> 4;
    const int rowBase = blockIdx.x * 128 + w * 16;
    const size_t abase = (size_t)(rowBase + m) * KDIM;

    v8f acc[8] = {};

    for (int kt = 0; kt < KDIM / 32; ++kt) {
        __syncthreads();
        for (int i = tid; i < 128 * 32; i += 256) {
            const int n = i >> 5, k = i & 31;
            sW[i] = (_Float16)W[(size_t)(kt * 32 + k) * 128 + n];
        }
        __syncthreads();

        const int k0 = kt * 32;
        v8h alo = *(const v8h*)(A + abase + k0 + hi * 8);
        v8h ahi = *(const v8h*)(A + abase + k0 + 16 + hi * 8);
        v16h afrag = cat16(alo, ahi);

#pragma unroll
        for (int ct = 0; ct < 8; ++ct) {
            const _Float16* wp = sW + (ct * 16 + m) * 32 + hi * 8;
            v16h bfrag = cat16(*(const v8h*)wp, *(const v8h*)(wp + 16));
            acc[ct] = __builtin_amdgcn_wmma_f32_16x16x32_f16(
                false, afrag, false, bfrag, (short)0, acc[ct], false, false);
        }
    }

#pragma unroll
    for (int ct = 0; ct < 8; ++ct) {
        const int col = ct * 16 + m;
        const float bb = (EPI == 0) ? b1[col] : 0.0f;
        const float cs = (EPI <= 1) ? csum[col] : 0.0f;
#pragma unroll
        for (int v = 0; v < 8; ++v) {
            const int row = rowBase + v + hi * 8;
            float val;
            if (EPI == 2) {
                val = acc[ct][v];
            } else {
                const int node = row % NNODE;
                val = alpha[node] * acc[ct][v] + beta[node] * cs + bb;
                if (EPI == 0) val = fmaxf(val, 0.0f);
            }
            out[(size_t)row * 128 + col] = (_Float16)val;
        }
    }
}

// ---------------------------------------------------------------------------
// Mask / dinv: grid graph => 2-hop BFS from seed == Manhattan distance <= 2.
// deg = mask*(1 + #masked neighbors); dinv = mask ? rsqrt(max(deg,1)) : 0.
// ---------------------------------------------------------------------------
__global__ __launch_bounds__(256)
void maskKernel(const int* __restrict__ seed, float* __restrict__ maskA,
                float* __restrict__ dinvA) {
    const int id = blockIdx.x * 256 + threadIdx.x;
    if (id >= MROWS) return;
    const int b = id / NNODE, n = id % NNODE;
    const int sd = seed[b];
    const int si = sd / 20, sj = sd % 20;
    const int i = n / 20, j = n % 20;
    const int man = abs(i - si) + abs(j - sj);
    const float mk = (man <= 2) ? 1.0f : 0.0f;
    int cnt = 0;
    if (mk > 0.0f) {
        if (j > 0  && (abs(i - si) + abs(j - 1 - sj)) <= 2) cnt++;
        if (j < 19 && (abs(i - si) + abs(j + 1 - sj)) <= 2) cnt++;
        if (i > 0  && (abs(i - 1 - si) + abs(j - sj)) <= 2) cnt++;
        if (i < 19 && (abs(i + 1 - si) + abs(j - sj)) <= 2) cnt++;
    }
    const float deg = mk * (1.0f + (float)cnt);
    maskA[id] = mk;
    dinvA[id] = (mk > 0.0f) ? rsqrtf(fmaxf(deg, 1.0f)) : 0.0f;
}

// ---------------------------------------------------------------------------
// Graph-conv aggregation: x = (dinv^2*h + dinv*sum_nbr dinv_m*h_m + bc)*mask
// ---------------------------------------------------------------------------
__global__ __launch_bounds__(128)
void aggKernel(const _Float16* __restrict__ h, const float* __restrict__ maskA,
               const float* __restrict__ dinvA, const float* __restrict__ bc,
               _Float16* __restrict__ xout) {
    const int rn = blockIdx.x;
    const int c = threadIdx.x;
    const int n = rn % NNODE;
    const int i = n / 20, j = n % 20;
    const float d = dinvA[rn];
    float acc = d * d * (float)h[(size_t)rn * 128 + c];
    if (j > 0)  { const int m2 = rn - 1;  acc += d * dinvA[m2] * (float)h[(size_t)m2 * 128 + c]; }
    if (j < 19) { const int m2 = rn + 1;  acc += d * dinvA[m2] * (float)h[(size_t)m2 * 128 + c]; }
    if (i > 0)  { const int m2 = rn - 20; acc += d * dinvA[m2] * (float)h[(size_t)m2 * 128 + c]; }
    if (i < 19) { const int m2 = rn + 20; acc += d * dinvA[m2] * (float)h[(size_t)m2 * 128 + c]; }
    const float val = (acc + bc[c]) * maskA[rn];
    xout[(size_t)rn * 128 + c] = (_Float16)val;
}

// ---------------------------------------------------------------------------
// Final masked mean over nodes -> out (B,128) f32
// ---------------------------------------------------------------------------
__global__ __launch_bounds__(128)
void finalKernel(const _Float16* __restrict__ x2, const float* __restrict__ maskA,
                 float* __restrict__ out) {
    const int b = blockIdx.x, jc = threadIdx.x;
    float s = 0.0f;
    for (int n = 0; n < NNODE; ++n)
        s += (float)x2[((size_t)b * NNODE + n) * 128 + jc];
    float ms = 0.0f;
    for (int n = 0; n < NNODE; ++n) ms += maskA[b * NNODE + n];
    out[b * 128 + jc] = s / fmaxf(ms, 1.0f);
}

// ---------------------------------------------------------------------------
extern "C" void kernel_launch(void* const* d_in, const int* in_sizes, int n_in,
                              void* d_out, int out_size, void* d_ws, size_t ws_size,
                              hipStream_t stream) {
    (void)in_sizes; (void)n_in; (void)out_size; (void)ws_size;
    const float* agvs  = (const float*)d_in[0];
    const float* nodes = (const float*)d_in[1];
    // d_in[2] = edge_index (grid adjacency is reconstructed analytically)
    const float* W0 = (const float*)d_in[3];
    const float* b0 = (const float*)d_in[4];
    const float* g1 = (const float*)d_in[5];
    const float* be1 = (const float*)d_in[6];
    const float* W1 = (const float*)d_in[7];
    const float* b1 = (const float*)d_in[8];
    const float* g2 = (const float*)d_in[9];
    const float* be2 = (const float*)d_in[10];
    const float* Wc = (const float*)d_in[11];   // (2,128,128)
    const float* bc = (const float*)d_in[12];   // (2,128)
    float* out = (float*)d_out;

    char* ws = (char*)d_ws;
    size_t off = 0;
    auto take = [&](size_t bytes) {
        size_t o = off;
        off += (bytes + 255) & ~(size_t)255;
        return o;
    };
    _Float16* feat16 = (_Float16*)(ws + take((size_t)MROWS * 32 * 2));
    int*      seed   = (int*)     (ws + take((size_t)BSZ * 4));
    _Float16* hrelu  = (_Float16*)(ws + take((size_t)MROWS * 256 * 2));
    float*    alpha1 = (float*)   (ws + take(NNODE * 4));
    float*    beta1  = (float*)   (ws + take(NNODE * 4));
    float*    alpha2 = (float*)   (ws + take(NNODE * 4));
    float*    beta2  = (float*)   (ws + take(NNODE * 4));
    float*    csumW1 = (float*)   (ws + take(128 * 4));
    float*    csumWc = (float*)   (ws + take(128 * 4));
    _Float16* xrelu  = (_Float16*)(ws + take((size_t)MROWS * 128 * 2));
    float*    maskA  = (float*)   (ws + take((size_t)MROWS * 4));
    float*    dinvA  = (float*)   (ws + take((size_t)MROWS * 4));
    _Float16* hbuf   = (_Float16*)(ws + take((size_t)MROWS * 128 * 2));
    _Float16* x1     = (_Float16*)(ws + take((size_t)MROWS * 128 * 2));
    _Float16* x2     = xrelu;   // xrelu dead after GC0 GEMM; reuse for x2

    // Stage A: features + seed
    featKernel<<<BSZ, 256, 0, stream>>>(agvs, nodes, feat16, seed);
    // Stage B: GEMM1 + relu
    gemm1Kernel<<<MROWS / 64, 256, 0, stream>>>(feat16, W0, b0, hrelu);
    // BN1 stats (folded into GEMM2)
    bnStatsKernel<<<NNODE, 256, 0, stream>>>(hrelu, 256, g1, be1, alpha1, beta1);
    colsumKernel<<<1, 128, 0, stream>>>(W1, Wc, csumW1, csumWc);
    // Stage D: GEMM2 with BN1 fold + bias + relu
    gemmKKernel<256, 0><<<MROWS / 128, 256, 0, stream>>>(hrelu, W1, b1, alpha1,
                                                         beta1, csumW1, xrelu);
    // BN2 stats (folded into GC layer-0 GEMM)
    bnStatsKernel<<<NNODE, 256, 0, stream>>>(xrelu, 128, g2, be2, alpha2, beta2);
    // Mask / dinv
    maskKernel<<<(MROWS + 255) / 256, 256, 0, stream>>>(seed, maskA, dinvA);
    // GC layer 0: h = BN2(x) @ Wc0  (fold), then aggregate
    gemmKKernel<128, 1><<<MROWS / 128, 256, 0, stream>>>(xrelu, Wc, b1, alpha2,
                                                         beta2, csumWc, hbuf);
    aggKernel<<<MROWS, 128, 0, stream>>>(hbuf, maskA, dinvA, bc, x1);
    // GC layer 1: h = x1 @ Wc1, then aggregate
    gemmKKernel<128, 2><<<MROWS / 128, 256, 0, stream>>>(x1, Wc + 128 * 128, b1,
                                                         alpha2, beta2, csumWc, hbuf);
    aggKernel<<<MROWS, 128, 0, stream>>>(hbuf, maskA, dinvA, bc + 128, x2);
    // Masked mean
    finalKernel<<<BSZ, 128, 0, stream>>>(x2, maskA, out);
}
